// BiasedAttention_13993003451056
// MI455X (gfx1250) — compile-verified, matrix operand order checked by reading, not run on hardware
//
#include <hip/hip_runtime.h>
#include <stdint.h>

// ---------------------------------------------------------------------------
// Types for CDNA5 WMMA (wave32, v_wmma_f32_16x16x32_bf16)
// ---------------------------------------------------------------------------
typedef __bf16 v16bf __attribute__((ext_vector_type(16)));
typedef float  v8f   __attribute__((ext_vector_type(8)));

union BF16Frag {
  v16bf bf;
  uint4 u4[2];
};

__device__ __forceinline__ unsigned short f32_to_bf16_rne(float f) {
  unsigned int u = __float_as_uint(f);
  u += 0x7fffu + ((u >> 16) & 1u);   // round to nearest even
  return (unsigned short)(u >> 16);
}

// Load a 16(row) x 32(K) bf16 fragment in the CDNA5 16-bit A/B register
// layout from a row-major matrix: lane = row (mod 16); lanes 0-15 hold
// K {0..7,16..23}, lanes 16-31 hold K {8..15,24..31}.  Two 16B loads/lane.
__device__ __forceinline__ v16bf load_frag(const unsigned short* base, int ld) {
  int lane = threadIdx.x & 31;
  int m  = lane & 15;
  int kb = (lane >> 4) << 3;          // 0 or 8
  const unsigned short* p = base + (size_t)m * ld + kb;
  BF16Frag f;
  f.u4[0] = *(const uint4*)(p);       // K = kb .. kb+7
  f.u4[1] = *(const uint4*)(p + 16);  // K = kb+16 .. kb+23
  return f.bf;
}

__device__ __forceinline__ v8f wmma_bf16(v16bf a, v16bf b, v8f c) {
  return __builtin_amdgcn_wmma_f32_16x16x32_bf16(
      /*neg_a=*/false, a, /*neg_b=*/false, b,
      /*c_mod=*/(short)0, c, /*reuse_a=*/false, /*reuse_b=*/false);
}

__device__ __forceinline__ v8f zero8() {
  v8f z = {0.f, 0.f, 0.f, 0.f, 0.f, 0.f, 0.f, 0.f};
  return z;
}

// 32x64x32 step: 2 A fragments x 4 B fragments -> 8 WMMAs
__device__ __forceinline__ void mma_step(const v16bf a[2], const v16bf b[4], v8f acc[8]) {
#pragma unroll
  for (int c = 0; c < 4; ++c) {
    acc[c]     = wmma_bf16(a[0], b[c], acc[c]);
    acc[4 + c] = wmma_bf16(a[1], b[c], acc[4 + c]);
  }
}

// Problem constants
#define BB     8
#define NN     1024
#define DIMM   1024
#define HH     16
#define DH     64
#define INNERX 1024
#define SCALE  0.125f   // 64^-0.5

// Load one K-step's worth of fragments (2 A, 4 B); ld fixed at 1024.
__device__ __forceinline__ void load_ab(const unsigned short* pA0, const unsigned short* pA1,
                                        const unsigned short* pB, int k0,
                                        v16bf a[2], v16bf b[4]) {
  a[0] = load_frag(pA0 + k0, DIMM);
  a[1] = load_frag(pA1 + k0, DIMM);
#pragma unroll
  for (int c = 0; c < 4; ++c) b[c] = load_frag(pB + (size_t)(c * 16) * DIMM + k0, DIMM);
}

// Ping-pong pipelined 32x64 GEMM core over K=1024: two buffer generations in
// distinct registers (no v_mov rotation), loads for step s+1 issued before the
// WMMAs of step s so waits are partial (loads complete in order).
__device__ __forceinline__ void gemm_32x64(const unsigned short* pA0,
                                           const unsigned short* pA1,
                                           const unsigned short* pB, v8f acc[8]) {
  v16bf a0[2], b0[4], a1[2], b1[4];
  load_ab(pA0, pA1, pB, 0, a0, b0);
  for (int s = 0; s < 30; s += 2) {
    load_ab(pA0, pA1, pB, (s + 1) * 32, a1, b1);
    __builtin_prefetch((const void*)(pB + s * 32 + 1024), 0, 1);
    mma_step(a0, b0, acc);
    load_ab(pA0, pA1, pB, (s + 2) * 32, a0, b0);
    mma_step(a1, b1, acc);
  }
  load_ab(pA0, pA1, pB, 31 * 32, a1, b1);
  mma_step(a0, b0, acc);
  mma_step(a1, b1, acc);
}

// ---------------------------------------------------------------------------
// K0: LayerNorm over last dim (1024) + cast to bf16.  One block per row.
// ---------------------------------------------------------------------------
__global__ void __launch_bounds__(256)
k_layernorm(const float* __restrict__ x, const float* __restrict__ gam,
            const float* __restrict__ bet, unsigned short* __restrict__ xn) {
  int row = blockIdx.x;                       // 0..B*N-1
  const float* xr = x + (size_t)row * DIMM;
  float v[4], s = 0.f, s2 = 0.f;
#pragma unroll
  for (int i = 0; i < 4; ++i) {
    v[i] = xr[threadIdx.x + 256 * i];
    s += v[i];
    s2 += v[i] * v[i];
  }
  __shared__ float r1[256], r2[256];
  r1[threadIdx.x] = s; r2[threadIdx.x] = s2;
  __syncthreads();
  for (int off = 128; off > 0; off >>= 1) {
    if (threadIdx.x < off) {
      r1[threadIdx.x] += r1[threadIdx.x + off];
      r2[threadIdx.x] += r2[threadIdx.x + off];
    }
    __syncthreads();
  }
  float mean = r1[0] * (1.f / DIMM);
  float var  = r2[0] * (1.f / DIMM) - mean * mean;
  float rstd = __frsqrt_rn(var + 1e-5f);
#pragma unroll
  for (int i = 0; i < 4; ++i) {
    int c = threadIdx.x + 256 * i;
    float y = (v[i] - mean) * rstd * gam[c] + bet[c];
    xn[(size_t)row * DIMM + c] = f32_to_bf16_rne(y);
  }
}

// ---------------------------------------------------------------------------
// K1: pack fp32 weight [K][N] into bf16 transposed [N][K] (B-fragment layout)
// ---------------------------------------------------------------------------
__global__ void k_pack_T(const float* __restrict__ w, unsigned short* __restrict__ wT,
                         int K, int Ncols) {
  int idx = blockIdx.x * blockDim.x + threadIdx.x;
  if (idx >= K * Ncols) return;
  int kk = idx / Ncols, n = idx % Ncols;
  wT[(size_t)n * K + kk] = f32_to_bf16_rne(w[idx]);
}

// ---------------------------------------------------------------------------
// K2: QKV GEMM [8192,1024] x [1024,3072], bf16 WMMA, 32x64 tile per wave,
// ping-pong pipelined.  Epilogue scatters q (pre-scaled), k (row-major) and
// v (transposed).
// ---------------------------------------------------------------------------
__global__ void __launch_bounds__(256)
k_qkv_gemm(const unsigned short* __restrict__ xn,
           const unsigned short* __restrict__ wT,
           unsigned short* __restrict__ q, unsigned short* __restrict__ kk,
           unsigned short* __restrict__ vT) {
  int wid  = (blockIdx.x * blockDim.x + threadIdx.x) >> 5;
  int lane = threadIdx.x & 31;
  int nt = wid % 48;                 // 48 * 64 = 3072 columns
  int mt = wid / 48;                 // 256 * 32 = 8192 rows
  const unsigned short* pA0 = xn + (size_t)(mt * 32) * DIMM;
  const unsigned short* pA1 = pA0 + 16 * DIMM;
  const unsigned short* pB  = wT + (size_t)(nt * 64) * DIMM;

  v8f acc[8];
#pragma unroll
  for (int i = 0; i < 8; ++i) acc[i] = zero8();

  gemm_32x64(pA0, pA1, pB, acc);

  int half = lane >> 4, cl = lane & 15;
#pragma unroll
  for (int sub = 0; sub < 2; ++sub) {
#pragma unroll
    for (int c = 0; c < 4; ++c) {
      int colg = nt * 64 + c * 16 + cl;
#pragma unroll
      for (int r = 0; r < 8; ++r) {
        int rowg = mt * 32 + sub * 16 + r + 8 * half;
        int b_ = rowg >> 10, n_ = rowg & 1023;
        float val = acc[sub * 4 + c][r];
        if (colg < 1024) {                   // Q, pre-scaled by d^-0.5
          int h = colg >> 6, d = colg & 63;
          q[(((size_t)b_ * HH + h) * NN + n_) * DH + d] = f32_to_bf16_rne(val * SCALE);
        } else if (colg < 2048) {            // K, row-major [n][d]
          int cc = colg - 1024; int h = cc >> 6, d = cc & 63;
          kk[(((size_t)b_ * HH + h) * NN + n_) * DH + d] = f32_to_bf16_rne(val);
        } else {                             // V, transposed [d][n]
          int cc = colg - 2048; int h = cc >> 6, d = cc & 63;
          vT[(((size_t)b_ * HH + h) * DH + d) * NN + n_] = f32_to_bf16_rne(val);
        }
      }
    }
  }
}

// ---------------------------------------------------------------------------
// K3: flash attention.  One wave owns 16 query rows of one (b,h); streams the
// 1024 keys in 32-column steps: S = qk^T (4 WMMA) + bias, online softmax,
// P staged through wave-private LDS into A-fragment layout, O += P x V
// (4 WMMA into a 16x64 f32 accumulator).  K fragments for step jp+1 and V
// fragments for step jp are issued around the S-WMMAs so the softmax
// VALU/trans work hides their latency.
// ---------------------------------------------------------------------------
__global__ void __launch_bounds__(256)
k_attention(const unsigned short* __restrict__ q,
            const unsigned short* __restrict__ kk,
            const unsigned short* __restrict__ vT,
            const float* __restrict__ bias,
            unsigned short* __restrict__ ao) {
  __shared__ unsigned short plds[8][16 * 32];    // 1 KB per wave P staging
  int lane = threadIdx.x & 31;
  int w    = threadIdx.x >> 5;
  int wid  = blockIdx.x * 8 + w;                 // 8192 waves
  int rb   = wid & 63;                           // query row-block
  int bh   = wid >> 6;                           // b*16 + h
  int h    = bh & 15;
  int half = lane >> 4, cl = lane & 15;

  const unsigned short* qp = q  + ((size_t)bh * NN + rb * 16) * DH;
  const unsigned short* kp = kk + (size_t)bh * NN * DH;
  const unsigned short* vp = vT + (size_t)bh * DH * NN;
  const float*          bp = bias + ((size_t)h * NN + rb * 16) * NN;

  v16bf qa0 = load_frag(qp, DH);         // d = 0..31
  v16bf qa1 = load_frag(qp + 32, DH);    // d = 32..63

  v8f o[4];
#pragma unroll
  for (int d = 0; d < 4; ++d) o[d] = zero8();
  float m_i[8], l_i[8];
#pragma unroll
  for (int r = 0; r < 8; ++r) { m_i[r] = -1e30f; l_i[r] = 0.f; }

  unsigned short* pl = plds[w];

  // prime the K-fragment pipeline (cols 0..31, both d-halves)
  v16bf kb[4];
  kb[0] = load_frag(kp, DH);
  kb[1] = load_frag(kp + 32, DH);
  kb[2] = load_frag(kp + (size_t)16 * DH, DH);
  kb[3] = load_frag(kp + (size_t)16 * DH + 32, DH);

  for (int jp = 0; jp < 32; ++jp) {
    int j0 = jp * 32;
    // V fragments for this step: issue early, consumed after softmax
    v16bf vb[4];
#pragma unroll
    for (int d = 0; d < 4; ++d)
      vb[d] = load_frag(vp + (size_t)(d * 16) * NN + j0, NN);

    // ---- S = q * k^T (16 x 32, two 16x16 C tiles) ----
    v8f s0 = zero8(), s1 = zero8();
    s0 = wmma_bf16(qa0, kb[0], s0);
    s0 = wmma_bf16(qa1, kb[1], s0);
    s1 = wmma_bf16(qa0, kb[2], s1);
    s1 = wmma_bf16(qa1, kb[3], s1);

    // prefetch next step's K fragments behind the S-WMMAs
    if (jp < 31) {
      int jn = j0 + 32;
      kb[0] = load_frag(kp + (size_t)jn * DH, DH);
      kb[1] = load_frag(kp + (size_t)jn * DH + 32, DH);
      kb[2] = load_frag(kp + (size_t)(jn + 16) * DH, DH);
      kb[3] = load_frag(kp + (size_t)(jn + 16) * DH + 32, DH);
    }

    float f0[8], f1[8], mnew[8];
#pragma unroll
    for (int r = 0; r < 8; ++r) {
      int row = r + 8 * half;
      f0[r] = s0[r] + bp[(size_t)row * NN + j0 + cl];
      f1[r] = s1[r] + bp[(size_t)row * NN + j0 + 16 + cl];
      // row max over the 16 lanes of this half-wave
      float mx = fmaxf(f0[r], f1[r]);
#pragma unroll
      for (int off = 8; off > 0; off >>= 1) mx = fmaxf(mx, __shfl_xor(mx, off, 32));
      mnew[r] = fmaxf(m_i[r], mx);
    }
#pragma unroll
    for (int r = 0; r < 8; ++r) {
      float alpha = __expf(m_i[r] - mnew[r]);
      f0[r] = __expf(f0[r] - mnew[r]);
      f1[r] = __expf(f1[r] - mnew[r]);
      float sum = f0[r] + f1[r];
#pragma unroll
      for (int off = 8; off > 0; off >>= 1) sum += __shfl_xor(sum, off, 32);
      l_i[r] = l_i[r] * alpha + sum;
      m_i[r] = mnew[r];
      o[0][r] *= alpha; o[1][r] *= alpha; o[2][r] *= alpha; o[3][r] *= alpha;
    }
    // ---- stage P (C layout -> row-major LDS) and reload as A fragment ----
#pragma unroll
    for (int r = 0; r < 8; ++r) {
      int row = r + 8 * half;
      pl[row * 32 + cl]      = f32_to_bf16_rne(f0[r]);
      pl[row * 32 + 16 + cl] = f32_to_bf16_rne(f1[r]);
    }
    v16bf pa = load_frag(pl, 32);        // wave-private: no barrier needed
    // ---- O += P x V  (V^T rows are B-fragment columns) ----
#pragma unroll
    for (int d = 0; d < 4; ++d) o[d] = wmma_bf16(pa, vb[d], o[d]);
  }
  // ---- normalize and emit bf16 attention output [b, n, h*64+d] ----
  int b_ = bh >> 4;
  float inv_l[8];
#pragma unroll
  for (int r = 0; r < 8; ++r) inv_l[r] = __frcp_rn(l_i[r]);
#pragma unroll
  for (int d = 0; d < 4; ++d) {
#pragma unroll
    for (int r = 0; r < 8; ++r) {
      int n_ = rb * 16 + r + 8 * half;
      float val = o[d][r] * inv_l[r];
      ao[((size_t)b_ * NN + n_) * INNERX + h * DH + d * 16 + cl] = f32_to_bf16_rne(val);
    }
  }
}

// ---------------------------------------------------------------------------
// K4: output projection [8192,1024] x [1024,1024] -> f32 out; 32x64 tile per
// wave, ping-pong pipelined like K2.
// ---------------------------------------------------------------------------
__global__ void __launch_bounds__(256)
k_out_gemm(const unsigned short* __restrict__ ao,
           const unsigned short* __restrict__ wT,
           float* __restrict__ out) {
  int wid  = (blockIdx.x * blockDim.x + threadIdx.x) >> 5;
  int lane = threadIdx.x & 31;
  int nt = wid & 15;                  // 16 * 64 = 1024 columns
  int mt = wid >> 4;                  // 256 * 32 = 8192 rows
  const unsigned short* pA0 = ao + (size_t)(mt * 32) * INNERX;
  const unsigned short* pA1 = pA0 + 16 * INNERX;
  const unsigned short* pB  = wT + (size_t)(nt * 64) * INNERX;

  v8f acc[8];
#pragma unroll
  for (int i = 0; i < 8; ++i) acc[i] = zero8();

  gemm_32x64(pA0, pA1, pB, acc);

  int half = lane >> 4, cl = lane & 15;
#pragma unroll
  for (int sub = 0; sub < 2; ++sub) {
#pragma unroll
    for (int c = 0; c < 4; ++c) {
#pragma unroll
      for (int r = 0; r < 8; ++r) {
        out[(size_t)(mt * 32 + sub * 16 + r + 8 * half) * DIMM + nt * 64 + c * 16 + cl] =
            acc[sub * 4 + c][r];
      }
    }
  }
}

// ---------------------------------------------------------------------------
// Host launcher
// ---------------------------------------------------------------------------
extern "C" void kernel_launch(void* const* d_in, const int* in_sizes, int n_in,
                              void* d_out, int out_size, void* d_ws, size_t ws_size,
                              hipStream_t stream) {
  (void)in_sizes; (void)n_in; (void)out_size; (void)ws_size;
  const float* x    = (const float*)d_in[0];
  const float* gam  = (const float*)d_in[1];
  const float* bet  = (const float*)d_in[2];
  const float* wqkv = (const float*)d_in[3];
  const float* wout = (const float*)d_in[4];
  const float* bias = (const float*)d_in[5];
  float* out = (float*)d_out;

  char* ws = (char*)d_ws;
  size_t off = 0;
  unsigned short* xn    = (unsigned short*)(ws + off); off += (size_t)BB * NN * DIMM * 2;      // 16 MB
  unsigned short* wqkvT = (unsigned short*)(ws + off); off += (size_t)3 * INNERX * DIMM * 2;   //  6 MB
  unsigned short* woutT = (unsigned short*)(ws + off); off += (size_t)DIMM * INNERX * 2;       //  2 MB
  unsigned short* q     = (unsigned short*)(ws + off); off += (size_t)BB * NN * INNERX * 2;    // 16 MB
  unsigned short* kk    = (unsigned short*)(ws + off); off += (size_t)BB * NN * INNERX * 2;    // 16 MB
  unsigned short* vT    = (unsigned short*)(ws + off); off += (size_t)BB * NN * INNERX * 2;    // 16 MB
  unsigned short* ao    = (unsigned short*)(ws + off); off += (size_t)BB * NN * INNERX * 2;    // 16 MB

  k_layernorm<<<BB * NN, 256, 0, stream>>>(x, gam, bet, xn);
  k_pack_T<<<(DIMM * 3 * INNERX + 255) / 256, 256, 0, stream>>>(wqkv, wqkvT, DIMM, 3 * INNERX);
  k_pack_T<<<(INNERX * DIMM + 255) / 256, 256, 0, stream>>>(wout, woutT, INNERX, DIMM);
  // 256 row tiles * 48 col groups = 12288 waves / 8 per block
  k_qkv_gemm<<<1536, 256, 0, stream>>>(xn, wqkvT, q, kk, vT);
  // 8 * 16 * 64 = 8192 waves / 8 per block
  k_attention<<<1024, 256, 0, stream>>>(q, kk, vT, bias, ao);
  // 256 row tiles * 16 col groups = 4096 waves / 8 per block
  k_out_gemm<<<512, 256, 0, stream>>>(ao, woutT, out);
}
